// Main_net_5609227288647
// MI455X (gfx1250) — matrix operand order repaired and checked by reading, not compiled, not run on hardware
//
#include <hip/hip_runtime.h>
#include <cstddef>

// ---------------------------------------------------------------------------
// Softsplat interpolation network for MI455X (gfx1250, wave32, WMMA).
// Convs: implicit-GEMM on v_wmma_f32_16x16x32_f16 with branch-free inner loop
// fed by pre-packed f16 operands (NHWC zero-padded activations + B-fragment
// packed weights), so the hot loop is pure vector loads + WMMA.
// ---------------------------------------------------------------------------

typedef __attribute__((ext_vector_type(16))) _Float16 v16h;
typedef __attribute__((ext_vector_type(8)))  _Float16 v8h;
typedef __attribute__((ext_vector_type(8)))  float    v8f;

#define EW_THREADS 256

// ---------------------------------------------------------------------------
// Weight packing: w[Cout][Cin][3][3] f32  ->  B fragments f16.
// Layout: wp[((oct*nChunks + chunk)*32 + lane)*16 + e]
//   chunk = tap*nC + cc   (tap-major K ordering, Cpad = 32*nC)
//   B 32x16 f16 fragment: lane L -> N = L%16 ; element e -> K = (e<8 ? e : e+8) + 8*(L>=16)
//   ci = cc*32 + K ; value = w[oc][ci][tap] (0 if padded)
// ---------------------------------------------------------------------------
__global__ void pack_weights_k(const float* __restrict__ w, _Float16* __restrict__ wp,
                               int Cin, int Cout, int nC, int tilesC)
{
    int nChunks = 9 * nC;
    int idx = blockIdx.x * blockDim.x + threadIdx.x;
    int total = tilesC * nChunks * 512;
    if (idx >= total) return;
    int e    = idx & 15;
    int lane = (idx >> 4) & 31;
    int rest = idx >> 9;
    int chunk = rest % nChunks;
    int oct   = rest / nChunks;
    int half = lane >> 4, l16 = lane & 15;
    int oc  = oct * 16 + l16;
    int tap = chunk / nC, cc = chunk % nC;
    int ci  = cc * 32 + (e < 8 ? e : e + 8) + 8 * half;
    float v = 0.f;
    if (ci < Cin && oc < Cout) v = w[((size_t)oc * Cin + ci) * 9 + tap];
    wp[idx] = (_Float16)v;
}

// ---------------------------------------------------------------------------
// Activation packing: NCHW f32 -> zero-padded NHWC f16.
// xp[((b*(Hin+2) + yy)*(Win+2) + xx)*Cpad + c], border rows/cols and padded
// channels are zero (gives SAME zero-padding for free, branch-free conv).
// ---------------------------------------------------------------------------
__global__ void pack_act_k(const float* __restrict__ x, _Float16* __restrict__ xp,
                           int Bn, int Cin, int Hin, int Win, int Cpad)
{
    int Hp = Hin + 2, Wp = Win + 2;
    int idx = blockIdx.x * blockDim.x + threadIdx.x;
    int total = Bn * Hp * Wp * Cpad;
    if (idx >= total) return;
    int c = idx % Cpad; int t = idx / Cpad;
    int xx = t % Wp; t /= Wp;
    int yy = t % Hp; int b = t / Hp;
    int iy = yy - 1, ix = xx - 1;
    float v = 0.f;
    if (c < Cin && iy >= 0 && iy < Hin && ix >= 0 && ix < Win)
        v = x[(((size_t)b * Cin + c) * Hin + iy) * Win + ix];
    xp[idx] = (_Float16)v;
}

// ---------------------------------------------------------------------------
// Implicit-GEMM 3x3 conv on packed operands. One wave32 computes a
// 16(spatial-x) x 16(oc) tile. Inner loop: 2 x 16B A loads + 1 x 32B B load
// + v_wmma_f32_16x16x32_f16, no branches, no divisions.
// A 16x32 f16 fragment: lane L -> M = L%16 ; element e ->
//   K = (e<8 ? e : e+8) + 8*(L>=16)  ==> bytes [16*half, +16) and [32+16*half, +16)
//   of the 64-byte 32-channel block -> two contiguous v8h loads.
// D 16x16 f32: element r -> M = r + 8*(L>=16), N = L%16.
// ---------------------------------------------------------------------------
__global__ void __launch_bounds__(128)
conv3x3_wmma(const _Float16* __restrict__ xp, const _Float16* __restrict__ wp,
             float* __restrict__ y,
             int Cpad, int Hp, int Wp,
             int Cout, int Hout, int Wout,
             int stride, int pad, int nC, int act,
             int yCtot, int yCoff, int tilesX, int tilesC, int nWaves)
{
    int wv = (int)((blockIdx.x * blockDim.x + threadIdx.x) >> 5);
    if (wv >= nWaves) return;                 // wave-uniform: EXEC stays all-1s
    int lane = threadIdx.x & 31;
    int half = lane >> 4;
    int l16  = lane & 15;

    int oct = wv % tilesC; int t = wv / tilesC;
    int oxt = t % tilesX;  t /= tilesX;
    int oy  = t % Hout;    int b = t / Hout;

    int nChunks = 9 * nC;
    v8f acc = {0.f, 0.f, 0.f, 0.f, 0.f, 0.f, 0.f, 0.f};

    int ox = oxt * 16 + l16;                  // A row (spatial)
    const _Float16* wbase = wp + (size_t)oct * nChunks * 512 + lane * 16;

    for (int tap = 0; tap < 9; ++tap) {
        int ky = tap / 3, kx = tap - ky * 3;
        int iy = oy * stride + ky - pad + 1;  // padded coords, always in range
        int ix = ox * stride + kx - pad + 1;
        const _Float16* ap = xp + (((size_t)b * Hp + iy) * Wp + ix) * Cpad + 8 * half;
        const _Float16* bp = wbase + (size_t)(tap * nC) * 512;
        for (int cc = 0; cc < nC; ++cc) {
            v8h a0 = *(const v8h*)ap;         // K in [8h, 8h+7]
            v8h a1 = *(const v8h*)(ap + 16);  // K in [16+8h, 16+8h+7]
            union { v16h v; v8h h[2]; } au;
            au.h[0] = a0; au.h[1] = a1;
            v16h Bm = *(const v16h*)bp;
            acc = __builtin_amdgcn_wmma_f32_16x16x32_f16(
                false, au.v, false, Bm, (short)0, acc, false, false);
            ap += 32; bp += 512;
        }
    }

    int oc = oct * 16 + l16;
    if (oc < Cout) {
#pragma unroll
        for (int r = 0; r < 8; ++r) {
            int oxx = oxt * 16 + r + half * 8;   // Wout is a multiple of 16 here
            float v = acc[r];
            if (act == 1)      v = fmaxf(v, 0.f);
            else if (act == 2) v = tanhf(v);
            y[(((size_t)b * yCtot + yCoff + oc) * Hout + oy) * Wout + oxx] = v;
        }
    }
}

// ---------------------------------------------------------------------------
// Half-pixel bilinear resize (jax.image.resize bilinear, antialias=False),
// edge-clamped. Optional scale and accumulate; channel-offset dst (concat).
// ---------------------------------------------------------------------------
__global__ void resize_bilinear_k(const float* __restrict__ src, float* __restrict__ dst,
                                  int Bn, int C, int Hs, int Ws, int Hd, int Wd,
                                  int dCtot, int dCoff, float scale, int accum)
{
    int idx = blockIdx.x * blockDim.x + threadIdx.x;
    int total = Bn * C * Hd * Wd;
    if (idx >= total) return;
    int xx = idx % Wd; int t = idx / Wd;
    int yy = t % Hd;   t /= Hd;
    int c  = t % C;    int b = t / C;

    float sx = (xx + 0.5f) * (float)Ws / (float)Wd - 0.5f;
    float sy = (yy + 0.5f) * (float)Hs / (float)Hd - 0.5f;
    int x0 = (int)floorf(sx), y0 = (int)floorf(sy);
    float fx = sx - (float)x0, fy = sy - (float)y0;
    int x0c = min(max(x0, 0), Ws - 1), x1c = min(max(x0 + 1, 0), Ws - 1);
    int y0c = min(max(y0, 0), Hs - 1), y1c = min(max(y0 + 1, 0), Hs - 1);

    const float* p = src + ((size_t)b * C + c) * Hs * Ws;
    float v00 = p[(size_t)y0c * Ws + x0c], v01 = p[(size_t)y0c * Ws + x1c];
    float v10 = p[(size_t)y1c * Ws + x0c], v11 = p[(size_t)y1c * Ws + x1c];
    float v = (v00 * (1.f - fx) + v01 * fx) * (1.f - fy)
            + (v10 * (1.f - fx) + v11 * fx) * fy;
    v *= scale;
    float* q = dst + (((size_t)b * dCtot + dCoff + c) * Hd + yy) * Wd + xx;
    if (accum) *q += v; else *q = v;
}

// Fused backwarp + mean|imgA - warp(imgB)| (zero-padded bilinear gather).
__global__ void warp_metric_k(const float* __restrict__ imgA, const float* __restrict__ imgB,
                              const float* __restrict__ flow, float* __restrict__ dst,
                              int Bn, int H, int W, int dCtot)
{
    int idx = blockIdx.x * blockDim.x + threadIdx.x;
    int HW = H * W;
    if (idx >= Bn * HW) return;
    int b = idx / HW, p = idx % HW;
    int yy = p / W, xx = p % W;

    float fx = flow[((size_t)b * 2 + 0) * HW + p];
    float fy = flow[((size_t)b * 2 + 1) * HW + p];
    float tx = xx + fx, ty = yy + fy;
    float x0f = floorf(tx), y0f = floorf(ty);
    int x0 = (int)x0f, y0 = (int)y0f;
    float dx = tx - x0f, dy = ty - y0f;

    float s = 0.f;
#pragma unroll
    for (int c = 0; c < 3; ++c) {
        const float* pb = imgB + ((size_t)b * 3 + c) * HW;
        float g = 0.f;
        if (x0   >= 0 && x0   < W && y0   >= 0 && y0   < H) g += pb[(size_t)y0 * W + x0] * (1.f - dx) * (1.f - dy);
        if (x0+1 >= 0 && x0+1 < W && y0   >= 0 && y0   < H) g += pb[(size_t)y0 * W + x0 + 1] * dx * (1.f - dy);
        if (x0   >= 0 && x0   < W && y0+1 >= 0 && y0+1 < H) g += pb[((size_t)y0 + 1) * W + x0] * (1.f - dx) * dy;
        if (x0+1 >= 0 && x0+1 < W && y0+1 >= 0 && y0+1 < H) g += pb[((size_t)y0 + 1) * W + x0 + 1] * dx * dy;
        s += fabsf(imgA[((size_t)b * 3 + c) * HW + p] - g);
    }
    dst[((size_t)b * dCtot + 0) * HW + p] = s * (1.f / 3.f);
}

__global__ void copy_channels_k(const float* __restrict__ src, float* __restrict__ dst,
                                int Bn, int C, int HW, int dCtot, int dCoff)
{
    int idx = blockIdx.x * blockDim.x + threadIdx.x;
    if (idx >= Bn * C * HW) return;
    int p = idx % HW; int t = idx / HW;
    int c = t % C;    int b = t / C;
    dst[(((size_t)b * dCtot) + dCoff + c) * HW + p] = src[((size_t)b * C + c) * HW + p];
}

// Softmax splatting scatter (4-corner bilinear atomics, exp(alpha*m) weights).
__global__ void splat_scatter_k(const float* __restrict__ inp, const float* __restrict__ flow,
                                const float* __restrict__ metric, const float* __restrict__ alpha,
                                float* __restrict__ acc,
                                int Bn, int C, int H, int W, float flowScale)
{
    int HW = H * W;
    int idx = blockIdx.x * blockDim.x + threadIdx.x;
    if (idx >= (C + 1) * Bn * HW) return;
    int c = idx / (Bn * HW);
    int r = idx % (Bn * HW);
    int b = r / HW, p = r % HW;
    int yy = p / W, xx = p % W;

    float fx = flow[((size_t)b * 2 + 0) * HW + p] * flowScale;
    float fy = flow[((size_t)b * 2 + 1) * HW + p] * flowScale;
    float z  = expf(alpha[0] * metric[(size_t)b * HW + p]);
    float val = (c < C) ? inp[((size_t)b * C + c) * HW + p] : 1.0f;
    val *= z;

    float tx = xx + fx, ty = yy + fy;
    float x0f = floorf(tx), y0f = floorf(ty);
    int x0 = (int)x0f, y0 = (int)y0f;
    float dx = tx - x0f, dy = ty - y0f;

    float* ap = acc + ((size_t)b * (C + 1) + c) * HW;
    if (x0   >= 0 && x0   < W && y0   >= 0 && y0   < H) atomicAdd(&ap[(size_t)y0 * W + x0],         val * (1.f - dx) * (1.f - dy));
    if (x0+1 >= 0 && x0+1 < W && y0   >= 0 && y0   < H) atomicAdd(&ap[(size_t)y0 * W + x0 + 1],     val * dx * (1.f - dy));
    if (x0   >= 0 && x0   < W && y0+1 >= 0 && y0+1 < H) atomicAdd(&ap[((size_t)y0 + 1) * W + x0],   val * (1.f - dx) * dy);
    if (x0+1 >= 0 && x0+1 < W && y0+1 >= 0 && y0+1 < H) atomicAdd(&ap[((size_t)y0 + 1) * W + x0+1], val * dx * dy);
}

__global__ void splat_normalize_k(const float* __restrict__ acc, float* __restrict__ dst,
                                  int Bn, int C, int HW, int dCtot, int dCoff)
{
    int idx = blockIdx.x * blockDim.x + threadIdx.x;
    if (idx >= Bn * C * HW) return;
    int c = idx / (Bn * HW);
    int r = idx % (Bn * HW);
    int b = r / HW, p = r % HW;
    float den = acc[((size_t)b * (C + 1) + C) * HW + p] + 1e-7f;
    dst[(((size_t)b * dCtot) + dCoff + c) * HW + p] =
        acc[((size_t)b * (C + 1) + c) * HW + p] / den;
}

// ---------------------------------------------------------------------------
// Host-side helpers
// ---------------------------------------------------------------------------
static inline int ew_blocks(long long n) { return (int)((n + EW_THREADS - 1) / EW_THREADS); }

static void run_conv(const float* x, const float* w, float* y,
                     int Bn, int Cin, int Hin, int Win,
                     int Cout, int Hout, int Wout,
                     int stride, int act, int yCtot, int yCoff,
                     _Float16* xpack, _Float16* wpack, hipStream_t s)
{
    int pad  = (stride == 1) ? 1 : 0;  // jax SAME: s1 -> 1/1, s2 even in -> 0/1
    int Cpad = ((Cin + 31) / 32) * 32;
    int nC   = Cpad / 32;
    int nChunks = 9 * nC;
    int tilesX = (Wout + 15) / 16;
    int tilesC = (Cout + 15) / 16;
    int Hp = Hin + 2, Wp = Win + 2;

    int wTotal = tilesC * nChunks * 512;
    pack_weights_k<<<ew_blocks(wTotal), EW_THREADS, 0, s>>>(w, wpack, Cin, Cout, nC, tilesC);
    long long aTotal = (long long)Bn * Hp * Wp * Cpad;
    pack_act_k<<<ew_blocks(aTotal), EW_THREADS, 0, s>>>(x, xpack, Bn, Cin, Hin, Win, Cpad);

    int nWaves = Bn * Hout * tilesX * tilesC;
    int blocks = (nWaves + 3) / 4;     // 128 threads = 4 waves per block
    conv3x3_wmma<<<blocks, 128, 0, s>>>(xpack, wpack, y, Cpad, Hp, Wp,
                                        Cout, Hout, Wout, stride, pad, nC, act,
                                        yCtot, yCoff, tilesX, tilesC, nWaves);
}

extern "C" void kernel_launch(void* const* d_in, const int* in_sizes, int n_in,
                              void* d_out, int out_size, void* d_ws, size_t ws_size,
                              hipStream_t stream)
{
    (void)in_sizes; (void)n_in; (void)out_size; (void)ws_size;
    const int Bn = 2, H = 512, W = 512;
    const int H2 = H / 2, W2 = W / 2, H4 = H / 4, W4 = W / 4;
    const size_t HW = (size_t)H * W, HW2 = (size_t)H2 * W2, HW4 = (size_t)H4 * W4;

    const float* img1     = (const float*)d_in[0];
    const float* img2     = (const float*)d_in[1];
    const float* fe_w1    = (const float*)d_in[2];
    const float* fe_w2    = (const float*)d_in[3];
    const float* fe_w3    = (const float*)d_in[4];
    const float* flow_w   = (const float*)d_in[5];
    const float* metric_w = (const float*)d_in[6];
    const float* gn_w1    = (const float*)d_in[7];
    const float* gn_w2    = (const float*)d_in[8];
    const float* gn_w3    = (const float*)d_in[9];
    const float* gn_out_w = (const float*)d_in[10];
    const float* alpha    = (const float*)d_in[11];
    float*       out      = (float*)d_out;

    // ---- workspace arena (256B aligned slots) ----
    char* base = (char*)d_ws;
    auto allocf = [&](size_t nelem) -> float* {
        float* p = (float*)base;
        base += ((nelem * sizeof(float) + 255) / 256) * 256;
        return p;
    };
    auto alloch = [&](size_t nelem) -> _Float16* {
        _Float16* p = (_Float16*)base;
        base += ((nelem * sizeof(_Float16) + 255) / 256) * 256;
        return p;
    };
    float* f1a = allocf((size_t)Bn * 32 * HW);
    float* f1b = allocf((size_t)Bn * 32 * HW);
    float* f2a = allocf((size_t)Bn * 64 * HW2);
    float* f2b = allocf((size_t)Bn * 64 * HW2);
    float* f3a = allocf((size_t)Bn * 96 * HW4);
    float* f3b = allocf((size_t)Bn * 96 * HW4);
    float* buf6 = allocf((size_t)Bn * 6 * HW4);
    float* fl12 = allocf((size_t)Bn * 2 * HW4);
    float* fl21 = allocf((size_t)Bn * 2 * HW4);
    float* fl12_s0 = allocf((size_t)Bn * 2 * HW);
    float* fl12_s1 = allocf((size_t)Bn * 2 * HW2);
    float* fl12_s2 = allocf((size_t)Bn * 2 * HW4);
    float* fl21_s0 = allocf((size_t)Bn * 2 * HW);
    float* fl21_s1 = allocf((size_t)Bn * 2 * HW2);
    float* fl21_s2 = allocf((size_t)Bn * 2 * HW4);
    float* metIn = allocf((size_t)Bn * 4 * HW);
    float* m12   = allocf((size_t)Bn * 1 * HW);
    float* m21   = allocf((size_t)Bn * 1 * HW);
    float* m12_1 = allocf((size_t)Bn * 1 * HW2);
    float* m21_1 = allocf((size_t)Bn * 1 * HW2);
    float* m12_2 = allocf((size_t)Bn * 1 * HW4);
    float* m21_2 = allocf((size_t)Bn * 1 * HW4);
    float* l1 = allocf((size_t)Bn * 70  * HW);
    float* l2 = allocf((size_t)Bn * 128 * HW2);
    float* l3 = allocf((size_t)Bn * 192 * HW4);
    size_t n_acc_wi  = (size_t)Bn * 4  * HW;
    size_t n_acc_wp0 = (size_t)Bn * 33 * HW;
    size_t n_acc_wp1 = (size_t)Bn * 65 * HW2;
    size_t n_acc_wp2 = (size_t)Bn * 97 * HW4;
    float* acc_wi  = allocf(n_acc_wi);
    float* acc_wp0 = allocf(n_acc_wp0);
    float* acc_wp1 = allocf(n_acc_wp1);
    float* acc_wp2 = allocf(n_acc_wp2);
    float* cl2 = allocf((size_t)Bn * 32 * HW2);
    float* cl3 = allocf((size_t)Bn * 32 * HW4);
    float* ybuf = allocf((size_t)Bn * 32 * HW);
    // shared packing scratch (convs are sequential on the stream):
    // max activation pack = l1: 2 * 514 * 514 * 96 f16
    _Float16* xpack = alloch((size_t)Bn * 514 * 514 * 96);
    _Float16* wpack = alloch(65536);

    // ---- feature pyramids (WMMA convs) ----
    run_conv(img1, fe_w1, f1a, Bn, 3, H, W, 32, H, W, 1, 1, 32, 0, xpack, wpack, stream);
    run_conv(f1a, fe_w2, f2a, Bn, 32, H, W, 64, H2, W2, 2, 1, 64, 0, xpack, wpack, stream);
    run_conv(f2a, fe_w3, f3a, Bn, 64, H2, W2, 96, H4, W4, 2, 1, 96, 0, xpack, wpack, stream);
    run_conv(img2, fe_w1, f1b, Bn, 3, H, W, 32, H, W, 1, 1, 32, 0, xpack, wpack, stream);
    run_conv(f1b, fe_w2, f2b, Bn, 32, H, W, 64, H2, W2, 2, 1, 64, 0, xpack, wpack, stream);
    run_conv(f2b, fe_w3, f3b, Bn, 64, H2, W2, 96, H4, W4, 2, 1, 96, 0, xpack, wpack, stream);

    // ---- flow nets: tanh(conv(resize(concat(a,b), H/4, W/4))) ----
    resize_bilinear_k<<<ew_blocks((long long)Bn*3*HW4), EW_THREADS, 0, stream>>>(
        img1, buf6, Bn, 3, H, W, H4, W4, 6, 0, 1.f, 0);
    resize_bilinear_k<<<ew_blocks((long long)Bn*3*HW4), EW_THREADS, 0, stream>>>(
        img2, buf6, Bn, 3, H, W, H4, W4, 6, 3, 1.f, 0);
    run_conv(buf6, flow_w, fl12, Bn, 6, H4, W4, 2, H4, W4, 1, 2, 2, 0, xpack, wpack, stream);
    resize_bilinear_k<<<ew_blocks((long long)Bn*3*HW4), EW_THREADS, 0, stream>>>(
        img2, buf6, Bn, 3, H, W, H4, W4, 6, 0, 1.f, 0);
    resize_bilinear_k<<<ew_blocks((long long)Bn*3*HW4), EW_THREADS, 0, stream>>>(
        img1, buf6, Bn, 3, H, W, H4, W4, 6, 3, 1.f, 0);
    run_conv(buf6, flow_w, fl21, Bn, 6, H4, W4, 2, H4, W4, 1, 2, 2, 0, xpack, wpack, stream);

    // ---- scaled flow pyramids: [20*up4, 10*up2, 5*same] ----
    resize_bilinear_k<<<ew_blocks((long long)Bn*2*HW ), EW_THREADS, 0, stream>>>(fl12, fl12_s0, Bn, 2, H4, W4, H,  W,  2, 0, 20.f, 0);
    resize_bilinear_k<<<ew_blocks((long long)Bn*2*HW2), EW_THREADS, 0, stream>>>(fl12, fl12_s1, Bn, 2, H4, W4, H2, W2, 2, 0, 10.f, 0);
    resize_bilinear_k<<<ew_blocks((long long)Bn*2*HW4), EW_THREADS, 0, stream>>>(fl12, fl12_s2, Bn, 2, H4, W4, H4, W4, 2, 0, 5.f,  0);
    resize_bilinear_k<<<ew_blocks((long long)Bn*2*HW ), EW_THREADS, 0, stream>>>(fl21, fl21_s0, Bn, 2, H4, W4, H,  W,  2, 0, 20.f, 0);
    resize_bilinear_k<<<ew_blocks((long long)Bn*2*HW2), EW_THREADS, 0, stream>>>(fl21, fl21_s1, Bn, 2, H4, W4, H2, W2, 2, 0, 10.f, 0);
    resize_bilinear_k<<<ew_blocks((long long)Bn*2*HW4), EW_THREADS, 0, stream>>>(fl21, fl21_s2, Bn, 2, H4, W4, H4, W4, 2, 0, 5.f,  0);

    // ---- metrics: conv(concat(mean|img - warp|, img)) + pyramids ----
    warp_metric_k<<<ew_blocks((long long)Bn*HW), EW_THREADS, 0, stream>>>(img1, img2, fl12_s0, metIn, Bn, H, W, 4);
    copy_channels_k<<<ew_blocks((long long)Bn*3*HW), EW_THREADS, 0, stream>>>(img1, metIn, Bn, 3, (int)HW, 4, 1);
    run_conv(metIn, metric_w, m12, Bn, 4, H, W, 1, H, W, 1, 0, 1, 0, xpack, wpack, stream);
    resize_bilinear_k<<<ew_blocks((long long)Bn*HW2), EW_THREADS, 0, stream>>>(m12, m12_1, Bn, 1, H, W, H2, W2, 1, 0, 1.f, 0);
    resize_bilinear_k<<<ew_blocks((long long)Bn*HW4), EW_THREADS, 0, stream>>>(m12, m12_2, Bn, 1, H, W, H4, W4, 1, 0, 1.f, 0);

    warp_metric_k<<<ew_blocks((long long)Bn*HW), EW_THREADS, 0, stream>>>(img2, img1, fl21_s0, metIn, Bn, H, W, 4);
    copy_channels_k<<<ew_blocks((long long)Bn*3*HW), EW_THREADS, 0, stream>>>(img2, metIn, Bn, 3, (int)HW, 4, 1);
    run_conv(metIn, metric_w, m21, Bn, 4, H, W, 1, H, W, 1, 0, 1, 0, xpack, wpack, stream);
    resize_bilinear_k<<<ew_blocks((long long)Bn*HW2), EW_THREADS, 0, stream>>>(m21, m21_1, Bn, 1, H, W, H2, W2, 1, 0, 1.f, 0);
    resize_bilinear_k<<<ew_blocks((long long)Bn*HW4), EW_THREADS, 0, stream>>>(m21, m21_2, Bn, 1, H, W, H4, W4, 1, 0, 1.f, 0);

    // ---- softmax splatting: direction 1->2 ----
    hipMemsetAsync(acc_wi,  0, n_acc_wi  * sizeof(float), stream);
    hipMemsetAsync(acc_wp0, 0, n_acc_wp0 * sizeof(float), stream);
    hipMemsetAsync(acc_wp1, 0, n_acc_wp1 * sizeof(float), stream);
    hipMemsetAsync(acc_wp2, 0, n_acc_wp2 * sizeof(float), stream);
    splat_scatter_k<<<ew_blocks((long long)4 *Bn*HW ), EW_THREADS, 0, stream>>>(img1, fl12_s0, m12,   alpha, acc_wi,  Bn, 3,  H,  W,  0.5f);
    splat_scatter_k<<<ew_blocks((long long)33*Bn*HW ), EW_THREADS, 0, stream>>>(f1a,  fl12_s0, m12,   alpha, acc_wp0, Bn, 32, H,  W,  0.5f);
    splat_scatter_k<<<ew_blocks((long long)65*Bn*HW2), EW_THREADS, 0, stream>>>(f2a,  fl12_s1, m12_1, alpha, acc_wp1, Bn, 64, H2, W2, 0.5f);
    splat_scatter_k<<<ew_blocks((long long)97*Bn*HW4), EW_THREADS, 0, stream>>>(f3a,  fl12_s2, m12_2, alpha, acc_wp2, Bn, 96, H4, W4, 0.5f);
    splat_normalize_k<<<ew_blocks((long long)Bn*3 *HW ), EW_THREADS, 0, stream>>>(acc_wi,  l1, Bn, 3,  (int)HW,  70,  0);
    splat_normalize_k<<<ew_blocks((long long)Bn*32*HW ), EW_THREADS, 0, stream>>>(acc_wp0, l1, Bn, 32, (int)HW,  70,  3);
    splat_normalize_k<<<ew_blocks((long long)Bn*64*HW2), EW_THREADS, 0, stream>>>(acc_wp1, l2, Bn, 64, (int)HW2, 128, 0);
    splat_normalize_k<<<ew_blocks((long long)Bn*96*HW4), EW_THREADS, 0, stream>>>(acc_wp2, l3, Bn, 96, (int)HW4, 192, 0);

    // ---- softmax splatting: direction 2->1 (reuse accumulators) ----
    hipMemsetAsync(acc_wi,  0, n_acc_wi  * sizeof(float), stream);
    hipMemsetAsync(acc_wp0, 0, n_acc_wp0 * sizeof(float), stream);
    hipMemsetAsync(acc_wp1, 0, n_acc_wp1 * sizeof(float), stream);
    hipMemsetAsync(acc_wp2, 0, n_acc_wp2 * sizeof(float), stream);
    splat_scatter_k<<<ew_blocks((long long)4 *Bn*HW ), EW_THREADS, 0, stream>>>(img2, fl21_s0, m21,   alpha, acc_wi,  Bn, 3,  H,  W,  0.5f);
    splat_scatter_k<<<ew_blocks((long long)33*Bn*HW ), EW_THREADS, 0, stream>>>(f1b,  fl21_s0, m21,   alpha, acc_wp0, Bn, 32, H,  W,  0.5f);
    splat_scatter_k<<<ew_blocks((long long)65*Bn*HW2), EW_THREADS, 0, stream>>>(f2b,  fl21_s1, m21_1, alpha, acc_wp1, Bn, 64, H2, W2, 0.5f);
    splat_scatter_k<<<ew_blocks((long long)97*Bn*HW4), EW_THREADS, 0, stream>>>(f3b,  fl21_s2, m21_2, alpha, acc_wp2, Bn, 96, H4, W4, 0.5f);
    splat_normalize_k<<<ew_blocks((long long)Bn*3 *HW ), EW_THREADS, 0, stream>>>(acc_wi,  l1, Bn, 3,  (int)HW,  70,  35);
    splat_normalize_k<<<ew_blocks((long long)Bn*32*HW ), EW_THREADS, 0, stream>>>(acc_wp0, l1, Bn, 32, (int)HW,  70,  38);
    splat_normalize_k<<<ew_blocks((long long)Bn*64*HW2), EW_THREADS, 0, stream>>>(acc_wp1, l2, Bn, 64, (int)HW2, 128, 64);
    splat_normalize_k<<<ew_blocks((long long)Bn*96*HW4), EW_THREADS, 0, stream>>>(acc_wp2, l3, Bn, 96, (int)HW4, 192, 96);

    // ---- grid net ----
    run_conv(l1, gn_w1, ybuf, Bn, 70,  H,  W,  32, H,  W,  1, 1, 32, 0, xpack, wpack, stream);
    run_conv(l2, gn_w2, cl2,  Bn, 128, H2, W2, 32, H2, W2, 1, 1, 32, 0, xpack, wpack, stream);
    run_conv(l3, gn_w3, cl3,  Bn, 192, H4, W4, 32, H4, W4, 1, 1, 32, 0, xpack, wpack, stream);
    resize_bilinear_k<<<ew_blocks((long long)Bn*32*HW), EW_THREADS, 0, stream>>>(cl2, ybuf, Bn, 32, H2, W2, H, W, 32, 0, 1.f, 1);
    resize_bilinear_k<<<ew_blocks((long long)Bn*32*HW), EW_THREADS, 0, stream>>>(cl3, ybuf, Bn, 32, H4, W4, H, W, 32, 0, 1.f, 1);
    run_conv(ybuf, gn_out_w, out, Bn, 32, H, W, 3, H, W, 1, 0, 3, 0, xpack, wpack, stream);
}
